// C2f_DualModal_MoE_51531017617523
// MI455X (gfx1250) — matrix operand order repaired and checked by reading, not compiled, not run on hardware
//
#include <hip/hip_runtime.h>
#include <hip/hip_bf16.h>

typedef __bf16 bf16;
typedef __attribute__((ext_vector_type(16))) __bf16 bf16x16;
typedef __attribute__((ext_vector_type(8)))  __bf16 bf16x8;
typedef __attribute__((ext_vector_type(8)))  float  f32x8;

#define NB    16      // batch
#define C1    256     // input channels
#define HW    6400    // 80*80
#define CC    384     // cat channels: a(128) | m(128) | moe(128)
#define KEXP  1152    // 128*9
#define LDA   40      // padded LDS K-stride (elements), 80B = 16B aligned

// fast SiLU: one v_rcp_f32 instead of the IEEE div sequence
__device__ __forceinline__ float silu_f(float v) {
    return v * __builtin_amdgcn_rcpf(1.f + __expf(-v));
}

__device__ __forceinline__ bf16x16 join8(bf16x8 lo, bf16x8 hi) {
    bf16x16 r;
#pragma unroll
    for (int i = 0; i < 8; ++i) { r[i] = lo[i]; r[i + 8] = hi[i]; }
    return r;
}

// A fragment: 16x32 bf16. lane L: row = L&15; lanes<16 hold K 0-7 & 16-23, lanes>=16 hold K 8-15 & 24-31.
__device__ __forceinline__ bf16x16 load_a_frag(const bf16* As, int row0, int lane) {
    int m  = lane & 15;
    int kb = (lane & 16) ? 8 : 0;
    const bf16* p = As + (row0 + m) * LDA + kb;
    bf16x8 lo = *(const bf16x8*)p;
    bf16x8 hi = *(const bf16x8*)(p + 16);
    return join8(lo, hi);
}

// B fragment: 32x16 bf16 stored [n][k]. lane L: col = L&15; lanes<16 hold K 0-7 & 16-23, lanes>=16 hold K 8-15 & 24-31.
__device__ __forceinline__ bf16x16 load_b_frag(const bf16* Bs, int col0, int lane) {
    int n  = col0 + (lane & 15);
    int kb = (lane & 16) ? 8 : 0;
    const bf16* p = Bs + n * LDA + kb;
    bf16x8 lo = *(const bf16x8*)p;
    bf16x8 hi = *(const bf16x8*)(p + 16);
    return join8(lo, hi);
}

#define WMMA_BF16(A, B, C) \
    __builtin_amdgcn_wmma_f32_16x16x32_bf16(false, (A), false, (B), (short)0, (C), false, false)

// Async global->LDS 16B copy (ASYNCcnt tracked). GVS mode:
// mem_addr = SADDR + VADDR(u32 byte offset); LDS dst = per-lane u32 byte address.
__device__ __forceinline__ void async_b128(unsigned lds_addr, unsigned long long gbase, unsigned voff) {
    asm volatile("global_load_async_to_lds_b128 %0, %1, %2"
                 :: "v"(lds_addr), "v"(voff), "s"(gbase) : "memory");
}
__device__ __forceinline__ void wait_async0() {
    asm volatile("s_wait_asynccnt 0x0" ::: "memory");
}
__device__ __forceinline__ unsigned lds_addr_of(const void* p) {
    return (unsigned)(uintptr_t)p;   // generic LDS pointer: addr[31:0] == LDS byte offset
}

// ---------------------------------------------------------------------------
// prep: bf16-convert weights (permute W_exp K to r*128+cin), zero pooled accum
// ---------------------------------------------------------------------------
__global__ void prep_kernel(const float* __restrict__ Wcv1, const float* __restrict__ Wexp,
                            const float* __restrict__ Wcv2,
                            bf16* __restrict__ wc1, bf16* __restrict__ wexp,
                            bf16* __restrict__ wc2, float* __restrict__ pooled) {
    int i = blockIdx.x * blockDim.x + threadIdx.x;
    int stride = gridDim.x * blockDim.x;
    for (int t = i; t < 256 * 256; t += stride) wc1[t] = (bf16)Wcv1[t];
    for (int t = i; t < 4 * 128 * KEXP; t += stride) {
        int e   = t / (128 * KEXP);
        int rem = t % (128 * KEXP);
        int co  = rem / KEXP;
        int kk  = rem % KEXP;
        int r   = kk / 128;           // 0..8 = (dh*3+dw)
        int ci  = kk % 128;
        wexp[t] = (bf16)Wexp[((size_t)(e * 128 + co) * 128 + ci) * 9 + r];
    }
    for (int t = i; t < 256 * CC; t += stride) wc2[t] = (bf16)Wcv2[t];
    for (int t = i; t < NB * 128; t += stride) pooled[t] = 0.f;
}

// ---------------------------------------------------------------------------
// cv1: 1x1 conv 256->256 + SiLU. GEMM M=256, K=256, N=102400.
// Double-buffered LDS ping-pong: async A copies + B staging of step s+1
// overlap the WMMAs of step s; one barrier per K-step.
// ---------------------------------------------------------------------------
__global__ __launch_bounds__(256) void cv1_kernel(const float* __restrict__ x,
                                                  const bf16* __restrict__ w1,
                                                  const float* __restrict__ b1,
                                                  bf16* __restrict__ cat,
                                                  float* __restrict__ pooled) {
    __shared__ bf16 As[2][128 * LDA];
    __shared__ bf16 Bs[2][64 * LDA];
    const int tid  = threadIdx.x;
    const int wave = tid >> 5, lane = tid & 31;
    const int gx = blockIdx.x * 64;          // global pixel base
    const int b  = gx / HW;
    const int p0 = gx % HW;
    const int c0 = blockIdx.y * 128;         // out-channel base

    const int nn = tid & 63;                 // B: pixel per thread
    const int kc = (tid >> 6) * 8;           // B: K chunk 0,8,16,24
    const int am = tid >> 1;                 // A: row 0..127
    const int ak = (tid & 1) * 16;           // A: K chunk

    const unsigned a_lds0 = lds_addr_of(&As[0][0]) + (unsigned)((am * LDA + ak) * 2);
    const unsigned long long w1b = (unsigned long long)w1;
    const float* src = x + (size_t)b * C1 * HW + p0 + nn;

    f32x8 acc0 = {}, acc1 = {}, acc2 = {}, acc3 = {};

    auto stage = [&](int s, int buf) {
        const int cin0 = s * 32;
        const unsigned avoff = (unsigned)(((c0 + am) * C1 + cin0 + ak) * 2);
        const unsigned alds  = a_lds0 + (unsigned)(buf * 128 * LDA * 2);
        async_b128(alds,      w1b, avoff);
        async_b128(alds + 16, w1b, avoff + 16);
        bf16x8 vals;
#pragma unroll
        for (int c = 0; c < 8; ++c)
            vals[c] = (bf16)src[(size_t)(cin0 + kc + c) * HW];
        *(bf16x8*)&Bs[buf][nn * LDA + kc] = vals;
        __builtin_prefetch(src + (size_t)(cin0 + kc + 32) * HW, 0, 3);
    };

    stage(0, 0);
    wait_async0();
    __syncthreads();

    const int S = C1 / 32;
#pragma unroll 2
    for (int s = 0; s < S; ++s) {
        const int cur = s & 1;
        if (s + 1 < S) stage(s + 1, cur ^ 1);
        bf16x16 a  = load_a_frag(As[cur], wave * 16, lane);
        bf16x16 f0 = load_b_frag(Bs[cur], 0,  lane);
        bf16x16 f1 = load_b_frag(Bs[cur], 16, lane);
        bf16x16 f2 = load_b_frag(Bs[cur], 32, lane);
        bf16x16 f3 = load_b_frag(Bs[cur], 48, lane);
        acc0 = WMMA_BF16(a, f0, acc0);
        acc1 = WMMA_BF16(a, f1, acc1);
        acc2 = WMMA_BF16(a, f2, acc2);
        acc3 = WMMA_BF16(a, f3, acc3);
        wait_async0();
        __syncthreads();
    }

    // epilogue: bias + SiLU, write bf16 into cat; pooled sums for m half
    const int m_lane = lane & 15;
    const int half   = lane >> 4;
#pragma unroll
    for (int v = 0; v < 8; ++v) {
        const int ch   = c0 + wave * 16 + half * 8 + v;   // 0..255 == cat channel
        const float bs = b1[ch];
        float o0 = silu_f(acc0[v] + bs);
        float o1 = silu_f(acc1[v] + bs);
        float o2 = silu_f(acc2[v] + bs);
        float o3 = silu_f(acc3[v] + bs);
        bf16* dst = cat + (size_t)b * CC * HW + (size_t)ch * HW + p0 + m_lane;
        dst[0]  = (bf16)o0;
        dst[16] = (bf16)o1;
        dst[32] = (bf16)o2;
        dst[48] = (bf16)o3;
        if (c0 == 128) {  // m half -> pooled accumulation
            float rs = o0 + o1 + o2 + o3;
#pragma unroll
            for (int off = 1; off < 16; off <<= 1) rs += __shfl_xor(rs, off, 32);
            if (m_lane == 0) atomicAdd(&pooled[b * 128 + (ch - 128)], rs);
        }
    }
}

// ---------------------------------------------------------------------------
// route: logits -> softmax -> top-2 -> normalized gate weights
// ---------------------------------------------------------------------------
__global__ void route_kernel(const float* __restrict__ pooled, const float* __restrict__ Wr,
                             const float* __restrict__ br, int* __restrict__ sel,
                             float* __restrict__ selw) {
    int b = threadIdx.x;
    if (b >= NB) return;
    float logit[4];
#pragma unroll
    for (int e = 0; e < 4; ++e) {
        float s = br[e];
        for (int c = 0; c < 128; ++c)
            s += (pooled[b * 128 + c] * (1.f / (float)HW)) * Wr[e * 128 + c];
        logit[e] = s;
    }
    float mx = logit[0];
#pragma unroll
    for (int e = 1; e < 4; ++e) mx = fmaxf(mx, logit[e]);
    float pr[4];
#pragma unroll
    for (int e = 0; e < 4; ++e) pr[e] = __expf(logit[e] - mx);
    int i0 = 0;
#pragma unroll
    for (int e = 1; e < 4; ++e) if (pr[e] > pr[i0]) i0 = e;
    int i1 = -1;
#pragma unroll
    for (int e = 0; e < 4; ++e) if (e != i0 && (i1 < 0 || pr[e] > pr[i1])) i1 = e;
    float w0 = pr[i0], w1 = pr[i1], ws = w0 + w1;
    sel[b * 2]      = i0;
    sel[b * 2 + 1]  = i1;
    selw[b * 2]     = w0 / ws;
    selw[b * 2 + 1] = w1 / ws;
}

// ---------------------------------------------------------------------------
// moe: top-2 expert 3x3 convs (implicit GEMM, K = r*128+cin), SiLU, gated sum.
// Double-buffered; batch b, 64 pixels x 64 out-ch; 8 waves 4(M) x 2(N).
// ---------------------------------------------------------------------------
__global__ __launch_bounds__(256) void moe_kernel(const bf16* __restrict__ cat_in,
                                                  const bf16* __restrict__ wexp,
                                                  const float* __restrict__ bexp,
                                                  const int* __restrict__ sel,
                                                  const float* __restrict__ selw,
                                                  bf16* __restrict__ cat) {
    __shared__ bf16 As[2][64 * LDA];
    __shared__ bf16 Bs[2][64 * LDA];
    const int tid  = threadIdx.x;
    const int wave = tid >> 5, lane = tid & 31;
    const int b  = blockIdx.z;
    const int p0 = blockIdx.x * 64;
    const int c0 = blockIdx.y * 64;
    const int mw = (wave & 3) * 16;
    const int nw = (wave >> 2) * 32;

    const bf16* mch = cat_in + (size_t)b * CC * HW + (size_t)128 * HW;  // m channels

    const int nn = tid & 63;                 // B: pixel per thread
    const int kc = (tid >> 6) * 8;
    const int p  = p0 + nn;
    const int h  = p / 80, w = p % 80;
    const int am = tid >> 2;                 // A: row 0..63
    const int ak = (tid & 3) * 8;

    const unsigned a_lds0 = lds_addr_of(&As[0][0]) + (unsigned)((am * LDA + ak) * 2);

    f32x8 moe0 = {}, moe1 = {};

    for (int ei = 0; ei < 2; ++ei) {
        const int   e  = sel[b * 2 + ei];
        const float wg = selw[b * 2 + ei];
        const unsigned long long web = (unsigned long long)(wexp + (size_t)e * 128 * KEXP);

        auto stage = [&](int s, int buf) {
            const int r    = s >> 2;               // tap index 0..8
            const int dh   = r / 3 - 1;
            const int dw   = r % 3 - 1;
            const int cin0 = (s & 3) * 32;
            async_b128(a_lds0 + (unsigned)(buf * 64 * LDA * 2), web,
                       (unsigned)(((c0 + am) * KEXP + r * 128 + cin0 + ak) * 2));
            const int hh = h + dh, ww = w + dw;
            const bool valid = ((unsigned)hh < 80u) && ((unsigned)ww < 80u);
            const bf16* mp = mch + (valid ? (hh * 80 + ww) : p);
            bf16x8 vals;
#pragma unroll
            for (int c = 0; c < 8; ++c)
                vals[c] = valid ? mp[(size_t)(cin0 + kc + c) * HW] : (bf16)0.f;
            *(bf16x8*)&Bs[buf][nn * LDA + kc] = vals;
        };

        f32x8 acc0 = {}, acc1 = {};
        stage(0, 0);
        wait_async0();
        __syncthreads();

        const int S = KEXP / 32;
#pragma unroll 2
        for (int s = 0; s < S; ++s) {
            const int cur = s & 1;
            if (s + 1 < S) stage(s + 1, cur ^ 1);
            bf16x16 a  = load_a_frag(As[cur], mw, lane);
            bf16x16 f0 = load_b_frag(Bs[cur], nw, lane);
            bf16x16 f1 = load_b_frag(Bs[cur], nw + 16, lane);
            acc0 = WMMA_BF16(a, f0, acc0);
            acc1 = WMMA_BF16(a, f1, acc1);
            wait_async0();
            __syncthreads();
        }

        const int half = lane >> 4;
#pragma unroll
        for (int v = 0; v < 8; ++v) {
            const int ch   = c0 + mw + half * 8 + v;
            const float bs = bexp[e * 128 + ch];
            moe0[v] += wg * silu_f(acc0[v] + bs);
            moe1[v] += wg * silu_f(acc1[v] + bs);
        }
    }

    const int m_lane = lane & 15;
    const int half   = lane >> 4;
    bf16* mo = cat + (size_t)b * CC * HW + (size_t)256 * HW;
#pragma unroll
    for (int v = 0; v < 8; ++v) {
        const int ch = c0 + mw + half * 8 + v;
        mo[(size_t)ch * HW + p0 + nw + m_lane]      = (bf16)moe0[v];
        mo[(size_t)ch * HW + p0 + nw + 16 + m_lane] = (bf16)moe1[v];
    }
}

// ---------------------------------------------------------------------------
// cv2: 1x1 conv 384->256 + SiLU over cat = [a|m|moe]. GEMM M=256, K=384.
// ---------------------------------------------------------------------------
__global__ __launch_bounds__(256) void cv2_kernel(const bf16* __restrict__ cat,
                                                  const bf16* __restrict__ w2,
                                                  const float* __restrict__ b2,
                                                  float* __restrict__ out) {
    __shared__ bf16 As[2][128 * LDA];
    __shared__ bf16 Bs[2][64 * LDA];
    const int tid  = threadIdx.x;
    const int wave = tid >> 5, lane = tid & 31;
    const int gx = blockIdx.x * 64;
    const int b  = gx / HW;
    const int p0 = gx % HW;
    const int c0 = blockIdx.y * 128;

    const int nn = tid & 63;
    const int kc = (tid >> 6) * 8;
    const int am = tid >> 1;
    const int ak = (tid & 1) * 16;

    const unsigned a_lds0 = lds_addr_of(&As[0][0]) + (unsigned)((am * LDA + ak) * 2);
    const unsigned long long w2b = (unsigned long long)w2;
    const bf16* cb = cat + (size_t)b * CC * HW + p0 + nn;

    f32x8 acc0 = {}, acc1 = {}, acc2 = {}, acc3 = {};

    auto stage = [&](int s, int buf) {
        const int k0 = s * 32;
        const unsigned avoff = (unsigned)(((c0 + am) * CC + k0 + ak) * 2);
        const unsigned alds  = a_lds0 + (unsigned)(buf * 128 * LDA * 2);
        async_b128(alds,      w2b, avoff);
        async_b128(alds + 16, w2b, avoff + 16);
        bf16x8 vals;
#pragma unroll
        for (int c = 0; c < 8; ++c)
            vals[c] = cb[(size_t)(k0 + kc + c) * HW];
        *(bf16x8*)&Bs[buf][nn * LDA + kc] = vals;
        __builtin_prefetch(cb + (size_t)(k0 + kc + 32) * HW, 0, 3);
    };

    stage(0, 0);
    wait_async0();
    __syncthreads();

    const int S = CC / 32;
#pragma unroll 2
    for (int s = 0; s < S; ++s) {
        const int cur = s & 1;
        if (s + 1 < S) stage(s + 1, cur ^ 1);
        bf16x16 a  = load_a_frag(As[cur], wave * 16, lane);
        bf16x16 f0 = load_b_frag(Bs[cur], 0,  lane);
        bf16x16 f1 = load_b_frag(Bs[cur], 16, lane);
        bf16x16 f2 = load_b_frag(Bs[cur], 32, lane);
        bf16x16 f3 = load_b_frag(Bs[cur], 48, lane);
        acc0 = WMMA_BF16(a, f0, acc0);
        acc1 = WMMA_BF16(a, f1, acc1);
        acc2 = WMMA_BF16(a, f2, acc2);
        acc3 = WMMA_BF16(a, f3, acc3);
        wait_async0();
        __syncthreads();
    }

    const int m_lane = lane & 15;
    const int half   = lane >> 4;
#pragma unroll
    for (int v = 0; v < 8; ++v) {
        const int ch   = c0 + wave * 16 + half * 8 + v;
        const float bs = b2[ch];
        float* dst = out + (size_t)b * 256 * HW + (size_t)ch * HW + p0 + m_lane;
        dst[0]  = silu_f(acc0[v] + bs);
        dst[16] = silu_f(acc1[v] + bs);
        dst[32] = silu_f(acc2[v] + bs);
        dst[48] = silu_f(acc3[v] + bs);
    }
}

// ---------------------------------------------------------------------------
extern "C" void kernel_launch(void* const* d_in, const int* in_sizes, int n_in,
                              void* d_out, int out_size, void* d_ws, size_t ws_size,
                              hipStream_t stream) {
    (void)in_sizes; (void)n_in; (void)out_size; (void)ws_size;
    const float* x    = (const float*)d_in[0];
    const float* Wcv1 = (const float*)d_in[1];
    const float* bcv1 = (const float*)d_in[2];
    const float* Wr   = (const float*)d_in[3];
    const float* br   = (const float*)d_in[4];
    const float* Wexp = (const float*)d_in[5];
    const float* bexp = (const float*)d_in[6];
    const float* Wcv2 = (const float*)d_in[7];
    const float* bcv2 = (const float*)d_in[8];
    float* out = (float*)d_out;

    char* ws = (char*)d_ws;
    bf16*  cat    = (bf16*)ws;  ws += (size_t)NB * CC * HW * sizeof(bf16);   // 78.6 MB
    bf16*  wc1b   = (bf16*)ws;  ws += (size_t)256 * C1 * sizeof(bf16);
    bf16*  wexpb  = (bf16*)ws;  ws += (size_t)4 * 128 * KEXP * sizeof(bf16);
    bf16*  wc2b   = (bf16*)ws;  ws += (size_t)256 * CC * sizeof(bf16);
    float* pooled = (float*)ws; ws += (size_t)NB * 128 * sizeof(float);
    int*   sel    = (int*)ws;   ws += (size_t)NB * 2 * sizeof(int);
    float* selw   = (float*)ws; ws += (size_t)NB * 2 * sizeof(float);

    prep_kernel<<<512, 256, 0, stream>>>(Wcv1, Wexp, Wcv2, wc1b, wexpb, wc2b, pooled);
    cv1_kernel<<<dim3(NB * HW / 64, 2), 256, 0, stream>>>(x, wc1b, bcv1, cat, pooled);
    route_kernel<<<1, 32, 0, stream>>>(pooled, Wr, br, sel, selw);
    moe_kernel<<<dim3(HW / 64, 2, NB), 256, 0, stream>>>(cat, wexpb, bexp, sel, selw, cat);
    cv2_kernel<<<dim3(NB * HW / 64, 2), 256, 0, stream>>>(cat, wc2b, bcv2, out);
}
